// pro_encoder_10857677325002
// MI455X (gfx1250) — compile-verified, matrix operand order checked by reading, not run on hardware
//
#include <hip/hip_runtime.h>
#include <hip/hip_bf16.h>
#include <math.h>

#define NN 50000
#define NE 1600000
#define FIN 33
#define FO 128
#define NH 2

typedef __attribute__((ext_vector_type(16))) __bf16 v16bf;
typedef __attribute__((ext_vector_type(8)))  __bf16 v8bf;
typedef __attribute__((ext_vector_type(8)))  float  v8f;

// ============================================================================
// WMMA GEMM path: operands pre-staged in bf16 with K padded to multiple of 32.
//   A  : [M x KP] bf16 row-major
//   Bt : [N x KP] bf16 row-major  (i.e., B transposed)
// Fragment layout (ISA 7.12.2, 16-bit 16x32): lane half (lane&16) selects the
// +0/+8 K-subgroup; per lane each fragment = two contiguous 16B chunks.
// Each wave computes a 16x64 C tile (4 accumulators share one A fragment).
// ============================================================================
template <int KP>
__global__ __launch_bounds__(32) void gemm_bf16_tile(const __bf16* __restrict__ A,
                                                     const __bf16* __restrict__ Bt,
                                                     float* __restrict__ C, int M,
                                                     int N) {
  const int lane = threadIdx.x;
  const int l15 = lane & 15;
  const int kh = (lane & 16) ? 8 : 0;
  const int m = blockIdx.x * 16 + l15;
  const int n0 = blockIdx.y * 64;

  const __bf16* arow = A + (size_t)m * KP + kh;
  const __bf16* brow = Bt + (size_t)(n0 + l15) * KP + kh;

  v8f acc[4] = {v8f{}, v8f{}, v8f{}, v8f{}};

#pragma unroll
  for (int k0 = 0; k0 < KP; k0 += 32) {
    if (k0 + 32 < KP) {
      __builtin_prefetch(arow + k0 + 32, 0, 1);  // global_prefetch_b8
    }
    const v8bf alo = *(const v8bf*)(arow + k0);
    const v8bf ahi = *(const v8bf*)(arow + k0 + 16);
    const v16bf a = __builtin_shufflevector(alo, ahi, 0, 1, 2, 3, 4, 5, 6, 7, 8,
                                            9, 10, 11, 12, 13, 14, 15);
#pragma unroll
    for (int t = 0; t < 4; ++t) {
      const __bf16* bp = brow + (size_t)t * 16 * KP + k0;
      const v8bf blo = *(const v8bf*)(bp);
      const v8bf bhi = *(const v8bf*)(bp + 16);
      const v16bf b = __builtin_shufflevector(blo, bhi, 0, 1, 2, 3, 4, 5, 6, 7,
                                              8, 9, 10, 11, 12, 13, 14, 15);
      acc[t] = __builtin_amdgcn_wmma_f32_16x16x32_bf16(
          /*neg_a=*/false, a, /*neg_b=*/false, b,
          /*c_mod=*/(short)0, acc[t], /*reuse_a=*/false, /*reuse_b=*/false);
    }
  }

  // C layout (ISA 7.12.2): lane half selects M+0/M+8, VGPR v = row offset.
  const int rbase = blockIdx.x * 16 + ((lane & 16) ? 8 : 0);
#pragma unroll
  for (int t = 0; t < 4; ++t) {
    float* cp = C + (size_t)rbase * N + n0 + t * 16 + l15;
#pragma unroll
    for (int v = 0; v < 8; ++v) {
      cp[(size_t)v * N] = acc[t][v];
    }
  }
}

// stage A: f32 [M x K] -> bf16 [M x KP], zero-padded
__global__ void stage_a_kernel(const float* __restrict__ A, __bf16* out, int M,
                               int K, int KP) {
  int i = blockIdx.x * blockDim.x + threadIdx.x;
  if (i >= M * KP) return;
  int m = i / KP, k = i - m * KP;
  out[i] = (k < K) ? (__bf16)A[(size_t)m * K + k] : (__bf16)0.0f;
}

// stage Bt: f32 [K x N] -> bf16 [N x KP] (transpose + pad)
__global__ void stage_bt_kernel(const float* __restrict__ B, __bf16* out, int K,
                                int N, int KP) {
  int i = blockIdx.x * blockDim.x + threadIdx.x;
  if (i >= N * KP) return;
  int n = i / KP, k = i - n * KP;
  out[i] = (k < K) ? (__bf16)B[(size_t)k * N + n] : (__bf16)0.0f;
}

// ---------- elementwise / graph kernels ----------
__global__ void fill_kernel(float* p, float v, int n) {
  int i = blockIdx.x * blockDim.x + threadIdx.x;
  if (i < n) p[i] = v;
}

__global__ void deg_edge_kernel(float* deg, const int* __restrict__ col,
                                const float* __restrict__ w, int E) {
  int e = blockIdx.x * blockDim.x + threadIdx.x;
  if (e < E) atomicAdd(&deg[col[e]], w[e]);
}

__global__ void dinv_kernel(const float* __restrict__ deg, float* dinv, int n) {
  int i = blockIdx.x * blockDim.x + threadIdx.x;
  if (i < n) {
    float d = deg[i];
    dinv[i] = d > 0.0f ? rsqrtf(d) : 0.0f;
  }
}

// agg[n][f] = dinv[n]^2 * xl[n][f]   (self-loop term of GCN)
__global__ void gcn_self_kernel(const float* __restrict__ xl,
                                const float* __restrict__ dinv, float* agg) {
  int i = blockIdx.x * blockDim.x + threadIdx.x;
  if (i < NN * FO) {
    int n = i >> 7;
    float di = dinv[n];
    agg[i] = di * di * xl[i];
  }
}

// one wave per edge, lane handles 4 features (float4 gather + f32 atomic scatter)
__global__ __launch_bounds__(256) void gcn_scatter_kernel(
    const float* __restrict__ xl, const float* __restrict__ dinv,
    const int* __restrict__ row, const int* __restrict__ col,
    const float* __restrict__ w, float* agg, int E) {
  int gid = blockIdx.x * blockDim.x + threadIdx.x;
  int e = gid >> 5, lane = gid & 31;
  if (e >= E) return;
  int r = row[e], c = col[e];
  float norm = dinv[r] * w[e] * dinv[c];
  const float4 v = *(const float4*)(xl + (size_t)r * FO + lane * 4);
  float* dst = agg + (size_t)c * FO + lane * 4;
  atomicAdd(dst + 0, norm * v.x);
  atomicAdd(dst + 1, norm * v.y);
  atomicAdd(dst + 2, norm * v.z);
  atomicAdd(dst + 3, norm * v.w);
}

// out[i] = act(in[i]*scale + b[i%128])
__global__ void bias_act_kernel(const float* __restrict__ in,
                                const float* __restrict__ b, float* out, int n,
                                float scale, int relu) {
  int i = blockIdx.x * blockDim.x + threadIdx.x;
  if (i < n) {
    float v = in[i] * scale + b[i & (FO - 1)];
    out[i] = relu ? fmaxf(v, 0.0f) : v;
  }
}

// per (node, head): attention dot products
__global__ void att_kernel(const float* __restrict__ xl,
                           const float* __restrict__ att_src,
                           const float* __restrict__ att_dst, float* asrc,
                           float* adst) {
  int i = blockIdx.x * blockDim.x + threadIdx.x;
  if (i >= NN * NH) return;
  int n = i >> 1, h = i & 1;
  const float* xp = xl + (size_t)n * (NH * FO) + h * FO;
  const float* as = att_src + h * FO;
  const float* ad = att_dst + h * FO;
  float ss = 0.0f, dd = 0.0f;
  for (int f = 0; f < FO; ++f) {
    float xv = xp[f];
    ss += xv * as[f];
    dd += xv * ad[f];
  }
  asrc[i] = ss;
  adst[i] = dd;
}

__device__ __forceinline__ unsigned fenc(float f) {
  unsigned u = __float_as_uint(f);
  return (u & 0x80000000u) ? ~u : (u | 0x80000000u);
}
__device__ __forceinline__ float fdec(unsigned u) {
  u = (u & 0x80000000u) ? (u & 0x7FFFFFFFu) : ~u;
  return __uint_as_float(u);
}
__device__ __forceinline__ float lrelu(float e) { return e > 0.0f ? e : 0.2f * e; }

// init per-dst running max with the self-loop logit
__global__ void minit_kernel(const float* __restrict__ asrc,
                             const float* __restrict__ adst, unsigned* menc) {
  int i = blockIdx.x * blockDim.x + threadIdx.x;
  if (i < NN * NH) menc[i] = fenc(lrelu(asrc[i] + adst[i]));
}

// per edge: compute logits (both heads), store, atomicMax into dst max
__global__ void emax_kernel(const float* __restrict__ asrc,
                            const float* __restrict__ adst,
                            const int* __restrict__ row,
                            const int* __restrict__ col, unsigned* menc,
                            float* ebuf, int E) {
  int e = blockIdx.x * blockDim.x + threadIdx.x;
  if (e >= E) return;
  int r = row[e], c = col[e];
#pragma unroll
  for (int h = 0; h < NH; ++h) {
    float v = lrelu(asrc[r * NH + h] + adst[c * NH + h]);
    ebuf[(size_t)e * NH + h] = v;
    atomicMax(&menc[c * NH + h], fenc(v));
  }
}

// decode max, init softmax denominator with self-loop term
__global__ void sinit_kernel(const float* __restrict__ asrc,
                             const float* __restrict__ adst,
                             const unsigned* __restrict__ menc, float* mf,
                             float* s) {
  int i = blockIdx.x * blockDim.x + threadIdx.x;
  if (i >= NN * NH) return;
  float mm = fdec(menc[i]);
  mf[i] = mm;
  s[i] = __expf(lrelu(asrc[i] + adst[i]) - mm);
}

// per (edge, head): p = exp(e - m[dst]); store p; s[dst] += p
__global__ void pexp_kernel(const int* __restrict__ col,
                            const float* __restrict__ mf, float* ebuf, float* s,
                            int E) {
  int i = blockIdx.x * blockDim.x + threadIdx.x;
  if (i >= E * NH) return;
  int e = i >> 1, h = i & 1;
  int c = col[e];
  float p = __expf(ebuf[i] - mf[c * NH + h]);
  ebuf[i] = p;
  atomicAdd(&s[c * NH + h], p);
}

// self-loop contribution to the aggregation (sum over heads)
__global__ void gat_self_kernel(const float* __restrict__ xl,
                                const float* __restrict__ asrc,
                                const float* __restrict__ adst,
                                const float* __restrict__ mf,
                                const float* __restrict__ s, float* agg) {
  int i = blockIdx.x * blockDim.x + threadIdx.x;
  if (i >= NN * FO) return;
  int n = i >> 7, f = i & (FO - 1);
  float acc = 0.0f;
#pragma unroll
  for (int h = 0; h < NH; ++h) {
    float alpha =
        __expf(lrelu(asrc[n * NH + h] + adst[n * NH + h]) - mf[n * NH + h]) /
        s[n * NH + h];
    acc += alpha * xl[(size_t)n * (NH * FO) + h * FO + f];
  }
  agg[i] = acc;
}

// one wave per edge: weighted gather of both heads, atomic scatter-add
__global__ __launch_bounds__(256) void gat_scatter_kernel(
    const float* __restrict__ xl, const int* __restrict__ row,
    const int* __restrict__ col, const float* __restrict__ ebuf,
    const float* __restrict__ s, float* agg, int E) {
  int gid = blockIdx.x * blockDim.x + threadIdx.x;
  int e = gid >> 5, lane = gid & 31;
  if (e >= E) return;
  int r = row[e], c = col[e];
  float a0 = ebuf[(size_t)e * NH + 0] / s[c * NH + 0];
  float a1 = ebuf[(size_t)e * NH + 1] / s[c * NH + 1];
  int f = lane * 4;
  const float4 v0 = *(const float4*)(xl + (size_t)r * (NH * FO) + f);
  const float4 v1 = *(const float4*)(xl + (size_t)r * (NH * FO) + FO + f);
  float* dst = agg + (size_t)c * FO + f;
  atomicAdd(dst + 0, a0 * v0.x + a1 * v1.x);
  atomicAdd(dst + 1, a0 * v0.y + a1 * v1.y);
  atomicAdd(dst + 2, a0 * v0.z + a1 * v1.z);
  atomicAdd(dst + 3, a0 * v0.w + a1 * v1.w);
}

// z = sigmoid(t1 + fc1b + t2 + fc2b + pro_bias); out = z*xt + (1-z)*x
__global__ void blend_kernel(const float* __restrict__ t1,
                             const float* __restrict__ fc1b,
                             const float* __restrict__ t2,
                             const float* __restrict__ fc2b,
                             const float* __restrict__ pb,
                             const float* __restrict__ xt,
                             const float* __restrict__ x, float* out) {
  int i = blockIdx.x * blockDim.x + threadIdx.x;
  if (i >= NN * FO) return;
  int f = i & (FO - 1);
  float v = t1[i] + fc1b[f] + t2[i] + fc2b[f] + pb[f];
  float z = 1.0f / (1.0f + __expf(-v));
  out[i] = z * xt[i] + (1.0f - z) * x[i];
}

// ---------- host orchestration ----------
static inline int cdiv(int a, int b) { return (a + b - 1) / b; }

// Full GEMM: C[MxN] = A[MxK] @ B[KxN], staging both operands to bf16 (K padded
// to KP). KP must be 64 or 128; M % 16 == 0; N % 64 == 0.
static void run_gemm(const float* A, const float* B, float* C, int M, int K,
                     int N, int KP, __bf16* Abf, __bf16* Btbf,
                     hipStream_t stream) {
  stage_a_kernel<<<cdiv(M * KP, 256), 256, 0, stream>>>(A, Abf, M, K, KP);
  stage_bt_kernel<<<cdiv(N * KP, 256), 256, 0, stream>>>(B, Btbf, K, N, KP);
  dim3 g(M / 16, N / 64);
  if (KP == 64) {
    gemm_bf16_tile<64><<<g, 32, 0, stream>>>(Abf, Btbf, C, M, N);
  } else {
    gemm_bf16_tile<128><<<g, 32, 0, stream>>>(Abf, Btbf, C, M, N);
  }
}

static void run_gat(const float* x, const float* W, const float* as,
                    const float* ad, const float* b, float* xl, float* asrc,
                    float* adst, unsigned* menc, float* mf, float* s,
                    float* ebuf, float* agg, float* xt, const int* row,
                    const int* col, __bf16* Abf, __bf16* Btbf, int relu,
                    hipStream_t stream) {
  run_gemm(x, W, xl, NN, FO, NH * FO, FO, Abf, Btbf, stream);
  att_kernel<<<cdiv(NN * NH, 256), 256, 0, stream>>>(xl, as, ad, asrc, adst);
  minit_kernel<<<cdiv(NN * NH, 256), 256, 0, stream>>>(asrc, adst, menc);
  emax_kernel<<<cdiv(NE, 256), 256, 0, stream>>>(asrc, adst, row, col, menc,
                                                 ebuf, NE);
  sinit_kernel<<<cdiv(NN * NH, 256), 256, 0, stream>>>(asrc, adst, menc, mf, s);
  pexp_kernel<<<cdiv(NE * NH, 256), 256, 0, stream>>>(col, mf, ebuf, s, NE);
  gat_self_kernel<<<cdiv(NN * FO, 256), 256, 0, stream>>>(xl, asrc, adst, mf, s,
                                                          agg);
  gat_scatter_kernel<<<cdiv(NE * 32, 256), 256, 0, stream>>>(xl, row, col, ebuf,
                                                             s, agg, NE);
  bias_act_kernel<<<cdiv(NN * FO, 256), 256, 0, stream>>>(
      agg, b, xt, NN * FO, 1.0f / (float)NH, relu);
}

extern "C" void kernel_launch(void* const* d_in, const int* in_sizes, int n_in,
                              void* d_out, int out_size, void* d_ws,
                              size_t ws_size, hipStream_t stream) {
  const float* target_x = (const float*)d_in[0];
  const int* eidx = (const int*)d_in[1];
  const float* tw = (const float*)d_in[2];
  const float* gcn_W = (const float*)d_in[3];
  const float* gcn_b = (const float*)d_in[4];
  const float* gat1_W = (const float*)d_in[5];
  const float* gat1_as = (const float*)d_in[6];
  const float* gat1_ad = (const float*)d_in[7];
  const float* gat1_b = (const float*)d_in[8];
  const float* gat2_W = (const float*)d_in[9];
  const float* gat2_as = (const float*)d_in[10];
  const float* gat2_ad = (const float*)d_in[11];
  const float* gat2_b = (const float*)d_in[12];
  const float* fc1_W = (const float*)d_in[13];
  const float* fc1_b = (const float*)d_in[14];
  const float* fc2_W = (const float*)d_in[15];
  const float* fc2_b = (const float*)d_in[16];
  const float* pro_bias = (const float*)d_in[17];

  const int* row = eidx;
  const int* col = eidx + NE;
  float* out = (float*)d_out;

  // carve workspace (all offsets multiples of 4 floats -> 16B aligned)
  float* ws = (float*)d_ws;
  size_t off = 0;
  float* deg = ws + off;  off += NN;
  float* dinv = ws + off; off += NN;
  float* xl = ws + off;   off += (size_t)NN * NH * FO;  // GAT xl; GCN uses NxFO
  float* xbuf = ws + off; off += (size_t)NN * FO;       // current features x
  float* xt = ws + off;   off += (size_t)NN * FO;
  float* agg = ws + off;  off += (size_t)NN * FO;       // reused as t1 in gating
  float* t2 = ws + off;   off += (size_t)NN * FO;
  float* asrc = ws + off; off += NN * NH;
  float* adst = ws + off; off += NN * NH;
  unsigned* menc = (unsigned*)(ws + off); off += NN * NH;
  float* mf = ws + off;   off += NN * NH;
  float* sbuf = ws + off; off += NN * NH;
  float* ebuf = ws + off; off += (size_t)NE * NH;
  __bf16* Abf = (__bf16*)(ws + off);  off += (size_t)NN * FO / 2;  // NNx128 bf16
  __bf16* Btbf = (__bf16*)(ws + off); off += (256 * 128) / 2;
  (void)ws_size; (void)n_in; (void)in_sizes; (void)out_size;

  // ---- layer 0: GCN + ReLU ----
  fill_kernel<<<cdiv(NN, 256), 256, 0, stream>>>(deg, 1.0f, NN);  // self-loops
  deg_edge_kernel<<<cdiv(NE, 256), 256, 0, stream>>>(deg, col, tw, NE);
  dinv_kernel<<<cdiv(NN, 256), 256, 0, stream>>>(deg, dinv, NN);
  run_gemm(target_x, gcn_W, xl, NN, FIN, FO, 64, Abf, Btbf, stream);
  gcn_self_kernel<<<cdiv(NN * FO, 256), 256, 0, stream>>>(xl, dinv, agg);
  gcn_scatter_kernel<<<cdiv(NE * 32, 256), 256, 0, stream>>>(xl, dinv, row, col,
                                                             tw, agg, NE);
  bias_act_kernel<<<cdiv(NN * FO, 256), 256, 0, stream>>>(agg, gcn_b, xbuf,
                                                          NN * FO, 1.0f, 1);

  // ---- layer 1: GAT + ReLU, gated blend ----
  run_gat(xbuf, gat1_W, gat1_as, gat1_ad, gat1_b, xl, asrc, adst, menc, mf,
          sbuf, ebuf, agg, xt, row, col, Abf, Btbf, /*relu=*/1, stream);
  run_gemm(xt, fc1_W, agg, NN, FO, FO, FO, Abf, Btbf, stream);   // t1 in agg
  run_gemm(xbuf, fc2_W, t2, NN, FO, FO, FO, Abf, Btbf, stream);
  blend_kernel<<<cdiv(NN * FO, 256), 256, 0, stream>>>(
      agg, fc1_b, t2, fc2_b, pro_bias, xt, xbuf, xbuf);  // x updated in place

  // ---- layer 2: GAT (no ReLU), gated blend -> d_out ----
  run_gat(xbuf, gat2_W, gat2_as, gat2_ad, gat2_b, xl, asrc, adst, menc, mf,
          sbuf, ebuf, agg, xt, row, col, Abf, Btbf, /*relu=*/0, stream);
  run_gemm(xt, fc1_W, agg, NN, FO, FO, FO, Abf, Btbf, stream);
  run_gemm(xbuf, fc2_W, t2, NN, FO, FO, FO, Abf, Btbf, stream);
  blend_kernel<<<cdiv(NN * FO, 256), 256, 0, stream>>>(
      agg, fc1_b, t2, fc2_b, pro_bias, xt, xbuf, out);
}
